// CustomMultiLossLayer_214748365194
// MI455X (gfx1250) — compile-verified, matrix operand order checked by reading, not run on hardware
//
#include <hip/hip_runtime.h>
#include <stdint.h>

typedef __attribute__((ext_vector_type(16))) _Float16 v16h;
typedef __attribute__((ext_vector_type(8)))  float    v8f;

// ------------------------------------------------------------- 32-bit RNG
// lowbias32 avalanche hash: 2x v_mul_lo_u32 + shifts/xors, no 64-bit carries.
__device__ __forceinline__ uint32_t hash32(uint32_t x) {
  x ^= x >> 16;
  x *= 0x7FEB352Du;
  x ^= x >> 15;
  x *= 0x846CA68Bu;
  x ^= x >> 16;
  return x;
}
__device__ __forceinline__ float u01(uint32_t v) {
  // uniform in (0,1), never 0 (safe for log)
  return (float)(v >> 8) * 0x1.0p-24f + 0x1.0p-25f;
}

// ---------------------------------------------------------------- init scratch
__global__ void __launch_bounds__(64)
zero_accum_kernel(float* acc) {
  if (threadIdx.x < 2) acc[threadIdx.x] = 0.0f;
}

// ------------------------------------------------- MC heteroscedastic CE loss
// Each thread owns one sample n and a strided slice of the T Monte-Carlo
// draws (t = chunk, chunk+S, ...). Per-sample terms are hoisted out of the
// T loop. Block reduction: LDS fold 256->32, then one v_wmma_f32_16x16x32_f16
// (A = partials, B = ones => D rows are row-sums; total = laneSum(0)+laneSum(16)),
// then a single atomicAdd per block.
__global__ void __launch_bounds__(256)
mc_hetero_ce_kernel(const float* __restrict__ tru, const float* __restrict__ pred,
                    int N, int T, int S, uint32_t seed, float* __restrict__ accum) {
  const int tid   = blockIdx.x * blockDim.x + threadIdx.x;
  const int total = N * S;
  float part = 0.0f;

  if (tid < total) {
    const int n     = tid % N;     // consecutive lanes -> consecutive samples
    const int chunk = tid / N;
    const float4* p4 = (const float4*)(pred + (size_t)n * 4);
    const float*  tr = tru + (size_t)n * 3;
    __builtin_prefetch(p4, 0, 3);  // global_prefetch_b8
    __builtin_prefetch(tr, 0, 3);

    const float4 pv = *p4;                          // one global_load_b128
    const float l0 = pv.x, l1 = pv.y, l2 = pv.z;
    const float scale = __expf(0.5f * pv.w);        // sqrt(exp(logvar))
    const float2 tv = *(const float2*)tr;           // global_load_b64
    const float t0 = tv.x, t1 = tv.y, t2 = tr[2];   // + global_load_b32
    const float tSum = t0 + t1 + t2;
    const float tDot = t0 * l0 + t1 * l1 + t2 * l2;

    // per-(t,n) counter; 4 independent 32-bit streams via lane offsets
    const uint32_t sbase = seed * 0x9E3779B9u;
    for (int t = chunk; t < T; t += S) {
      const uint32_t ctr = sbase + (uint32_t)(t * N + n) * 4u;
      const uint32_t r0 = hash32(ctr + 0u);
      const uint32_t r1 = hash32(ctr + 1u);
      const uint32_t r2 = hash32(ctr + 2u);
      const uint32_t r3 = hash32(ctr + 3u);

      // Box-Muller: 3 standard normals per draw
      const float rad0 = sqrtf(-2.0f * __logf(u01(r0)));
      float sn0, cs0; __sincosf(6.28318530718f * u01(r1), &sn0, &cs0);
      const float e0 = rad0 * cs0, e1 = rad0 * sn0;
      const float rad1 = sqrtf(-2.0f * __logf(u01(r2)));
      float sn1, cs1; __sincosf(6.28318530718f * u01(r3), &sn1, &cs1);
      const float e2 = rad1 * cs1;

      // noisy logits + 3-way log-sum-exp
      const float a = fmaf(scale, e0, l0);
      const float b = fmaf(scale, e1, l1);
      const float c = fmaf(scale, e2, l2);
      const float m = fmaxf(a, fmaxf(b, c));
      const float lse = m + __logf(__expf(a - m) + __expf(b - m) + __expf(c - m));

      // ce = -sum(true * log_softmax(noisy))
      const float eDot = t0 * e0 + t1 * e1 + t2 * e2;
      part += tSum * lse - fmaf(scale, eDot, tDot);
    }
  }

  __shared__ float red[256];
  red[threadIdx.x] = part;
  __syncthreads();

  if (threadIdx.x < 32) {            // whole wave 0 enters: EXEC all-ones for WMMA
    float s = 0.0f;
#pragma unroll
    for (int j = 0; j < 8; ++j) s += red[threadIdx.x + 32 * j];

    v16h av = {};                    // one A slot per lane -> bijective (M,K) coverage
    av[0] = (_Float16)s;
    v16h ones;
#pragma unroll
    for (int i = 0; i < 16; ++i) ones[i] = (_Float16)1.0f;
    v8f cacc = {};
    // D[m][n] = rowsum_A(m):   (neg_a, A, neg_b, B, c_mod, C, reuse_a, reuse_b)
    cacc = __builtin_amdgcn_wmma_f32_16x16x32_f16(
        false, av, false, ones, (short)0, cacc, false, false);

    float dsum = 0.0f;
#pragma unroll
    for (int i = 0; i < 8; ++i) dsum += cacc[i];   // lane L: rows 0-7 (L<16) / 8-15
    const float other = __shfl(dsum, 16, 32);
    if (threadIdx.x == 0) atomicAdd(accum, dsum + other);
  }
}

// ---------------------------------------------------------------- finalize
__global__ void __launch_bounds__(64)
finalize_kernel(const float* __restrict__ acc, const float* __restrict__ log_vars,
                const float* __restrict__ w_img, const float* __restrict__ w_cls,
                float inv_img, float inv_cls, float* __restrict__ out) {
  if (threadIdx.x == 0 && blockIdx.x == 0) {
    const float l_img = acc[0] * inv_img *
                        ((w_img[0] + w_img[1] + w_img[2]) * (1.0f / 3.0f));
    const float l_cls = acc[1] * inv_cls *
                        ((w_cls[0] + w_cls[1] + w_cls[2]) * (1.0f / 3.0f));
    const float lv0 = log_vars[0], lv1 = log_vars[1];
    out[0] = __expf(-lv0) * l_img + lv0 + __expf(-lv1) * l_cls + lv1;
  }
}

// ---------------------------------------------------------------- launcher
extern "C" void kernel_launch(void* const* d_in, const int* in_sizes, int n_in,
                              void* d_out, int out_size, void* d_ws, size_t ws_size,
                              hipStream_t stream) {
  const float* true_img = (const float*)d_in[0];
  const float* pred_img = (const float*)d_in[1];
  const float* true_cls = (const float*)d_in[2];
  const float* pred_cls = (const float*)d_in[3];
  const float* log_vars = (const float*)d_in[4];
  const float* w_img    = (const float*)d_in[5];
  const float* w_cls    = (const float*)d_in[6];
  float* out = (float*)d_out;
  float* acc = (float*)d_ws;         // acc[0] = img sum, acc[1] = cls sum

  const int C     = 3;
  const int N_img = in_sizes[0] / C;   // 65536
  const int N_cls = in_sizes[2] / C;   // 4
  const int T     = 500;

  zero_accum_kernel<<<1, 64, 0, stream>>>(acc);

  const int S_img = 4;                 // 4 T-chunks per sample -> 8192 waves
  const int blocks_img = (N_img * S_img + 255) / 256;
  mc_hetero_ce_kernel<<<blocks_img, 256, 0, stream>>>(
      true_img, pred_img, N_img, T, S_img, 123u, acc + 0);

  const int S_cls = 64;                // tiny branch: one block
  const int blocks_cls = (N_cls * S_cls + 255) / 256;
  mc_hetero_ce_kernel<<<blocks_cls, 256, 0, stream>>>(
      true_cls, pred_cls, N_cls, T, S_cls, 456u, acc + 1);

  finalize_kernel<<<1, 64, 0, stream>>>(
      acc, log_vars, w_img, w_cls,
      1.0f / ((float)T * (float)N_img),
      1.0f / ((float)T * (float)N_cls), out);
}
